// MultiHeadAttention_11562051960992
// MI455X (gfx1250) — compile-verified, matrix operand order checked by reading, not run on hardware
//
#include <hip/hip_runtime.h>

// ---------------- problem constants ----------------
#define Bb   8
#define Nn   1024
#define Cc   768
#define Hh   12
#define Dd   64
#define C3   2304   // 3*C

typedef __attribute__((ext_vector_type(16))) _Float16 v16h;
typedef __attribute__((ext_vector_type(8)))  _Float16 v8h;
typedef __attribute__((ext_vector_type(8)))  float    v8f;

// workspace offsets in _Float16 units
#define XH_OFF     0u                       // [B*N, C]           6291456
#define WQKVT_OFF  6291456u                 // [3C, C] (W^T)      1769472
#define WPROJT_OFF 8060928u                 // [C, C]  (W^T)       589824
#define QH_OFF     8650752u                 // [B,H,N,D]          6291456
#define KH_OFF     14942208u                // [B,H,N,D]          6291456
#define VT_OFF     21233664u                // [B,H,D,N]          6291456
#define CTXH_OFF   27525120u                // [B,N,C]            6291456

#define WMMA_F16(a, b, c) \
  __builtin_amdgcn_wmma_f32_16x16x32_f16(false, (a), false, (b), (short)0, (c), false, false)

#define WAIT_ASYNC(n) asm volatile("s_wait_asynccnt " #n ::: "memory")

__device__ __forceinline__ int lane_id() { return (int)(threadIdx.x & 31u); }

// ---- async stage: copy one 64x32-half B^T chunk (4KB) global -> LDS ----
// 256 threads, one global_load_async_to_lds_b128 each (ASYNCcnt-tracked).
__device__ __forceinline__ void stage_async(const _Float16* __restrict__ g, int ld, int kb,
                                            _Float16* lbuf) {
  int tid = (int)(threadIdx.x & 255u);
  int row = tid >> 2;          // 0..63
  int seg = (tid & 3) << 3;    // 0,8,16,24 halves
  const _Float16* gp = g + (size_t)row * ld + kb + seg;
  unsigned loff = (unsigned)(uintptr_t)(lbuf + row * 32 + seg);  // LDS byte offset (low 32b)
  unsigned long long ga = (unsigned long long)(uintptr_t)gp;
  asm volatile("global_load_async_to_lds_b128 %0, %1, off"
               :: "v"(loff), "v"(ga) : "memory");
}

// ---- A fragment (16x32 f16, row-major, leading dim ld) from global ----
// CDNA5 16-bit A layout: lanes 0-15: K in [0..8)+[16..24); lanes 16-31: [8..16)+[24..32).
__device__ __forceinline__ v16h load_frag_h(const _Float16* __restrict__ base, int ld, int kb) {
  int lane = lane_id();
  int r    = lane & 15;
  int sel  = lane >> 4;
  const _Float16* p = base + (size_t)r * ld + kb + sel * 8;
  v8h lo = *(const v8h*)(p);
  v8h hi = *(const v8h*)(p + 16);
  v16h out;
#pragma unroll
  for (int i = 0; i < 8; ++i) { out[i] = lo[i]; out[i + 8] = hi[i]; }
  return out;
}

// ---- A fragment sourced from f32 memory (on-the-fly f32 -> f16 convert) ----
__device__ __forceinline__ v16h load_frag_f32(const float* __restrict__ base, int ld, int kb) {
  int lane = lane_id();
  int r    = lane & 15;
  int sel  = lane >> 4;
  const float* p = base + (size_t)r * ld + kb + sel * 8;
  float lo[8], hi[8];
  *(float4*)(lo + 0) = *(const float4*)(p + 0);
  *(float4*)(lo + 4) = *(const float4*)(p + 4);
  *(float4*)(hi + 0) = *(const float4*)(p + 16);
  *(float4*)(hi + 4) = *(const float4*)(p + 20);
  v16h out;
#pragma unroll
  for (int i = 0; i < 8; ++i) { out[i] = (_Float16)lo[i]; out[i + 8] = (_Float16)hi[i]; }
  return out;
}

// ---- B fragment from a staged 64x32-half LDS chunk (rows jrow..jrow+15) ----
__device__ __forceinline__ v16h lds_frag(const _Float16* lbuf, int jrow) {
  int lane = lane_id();
  int r    = jrow + (lane & 15);
  int sel  = lane >> 4;
  const _Float16* p = lbuf + r * 32 + sel * 8;
  v8h lo = *(const v8h*)(p);
  v8h hi = *(const v8h*)(p + 16);
  v16h out;
#pragma unroll
  for (int i = 0; i < 8; ++i) { out[i] = lo[i]; out[i + 8] = hi[i]; }
  return out;
}

// ---------------- conversion kernels ----------------
__global__ __launch_bounds__(256) void cvt_f32_to_f16(const float* __restrict__ src,
                                                      _Float16* __restrict__ dst, int n) {
  int i = blockIdx.x * 256 + threadIdx.x;
  if (i < n) dst[i] = (_Float16)src[i];
}

// src: [rows_in, cols_out] f32 ; dst: [cols_out, rows_in] f16 (transposed)
__global__ __launch_bounds__(256) void cvt_transpose_f16(const float* __restrict__ src,
                                                         _Float16* __restrict__ dst,
                                                         int rows_in, int cols_out) {
  int i = blockIdx.x * 256 + threadIdx.x;
  int total = rows_in * cols_out;
  if (i < total) {
    int o = i / rows_in;
    int c = i - o * rows_in;
    dst[i] = (_Float16)src[(size_t)c * cols_out + o];
  }
}

// Core pipelined loop body shared by all GEMMs (macro to keep accs in scalars).
// Each wave: 2 M tiles x 64 N cols; 8 WMMAs per staged 4KB chunk.
#define GEMM_PIPELINE(NIT, LD, LOAD_A0, LOAD_A1)                                  \
  _Float16* cur = sB[0];                                                          \
  _Float16* nxt = sB[1];                                                          \
  stage_async(Bp, (LD), 0, cur);                                                  \
  v8f c00{0}, c01{0}, c02{0}, c03{0}, c10{0}, c11{0}, c12{0}, c13{0};             \
  _Pragma("unroll 1")                                                             \
  for (int it = 0; it < (NIT); ++it) {                                            \
    int kb = it * 32;                                                             \
    v16h a0 = (LOAD_A0);                                                          \
    v16h a1 = (LOAD_A1);                                                          \
    int kbn = (it + 1 == (NIT)) ? 0 : kb + 32;                                    \
    stage_async(Bp, (LD), kbn, nxt);                                              \
    WAIT_ASYNC(0x1);                                                              \
    __syncthreads();                                                              \
    v16h b0 = lds_frag(cur, 0);                                                   \
    v16h b1 = lds_frag(cur, 16);                                                  \
    v16h b2 = lds_frag(cur, 32);                                                  \
    v16h b3 = lds_frag(cur, 48);                                                  \
    c00 = WMMA_F16(a0, b0, c00); c10 = WMMA_F16(a1, b0, c10);                     \
    c01 = WMMA_F16(a0, b1, c01); c11 = WMMA_F16(a1, b1, c11);                     \
    c02 = WMMA_F16(a0, b2, c02); c12 = WMMA_F16(a1, b2, c12);                     \
    c03 = WMMA_F16(a0, b3, c03); c13 = WMMA_F16(a1, b3, c13);                     \
    __syncthreads();                                                              \
    _Float16* t = cur; cur = nxt; nxt = t;                                        \
  }                                                                               \
  v8f accs[2][4] = {{c00, c01, c02, c03}, {c10, c11, c12, c13}};

// ---------------- QKV GEMM: [B*N,C] x [C,3C], async-LDS pipelined ----------------
__global__ __launch_bounds__(256) void qkv_gemm(const _Float16* __restrict__ xh,
                                                const _Float16* __restrict__ wT,
                                                _Float16* __restrict__ qh,
                                                _Float16* __restrict__ kh,
                                                _Float16* __restrict__ vT) {
  __shared__ _Float16 sB[2][64 * 32];
  int wid = (int)(threadIdx.x >> 5);
  int nt  = blockIdx.x % 36;                 // 36 groups of 64 output cols
  int mtg = blockIdx.x / 36;                 // 32 groups of 16 M tiles
  int mt0 = mtg * 16 + wid * 2;
  const _Float16* A  = xh + (size_t)mt0 * 16 * Cc;
  const _Float16* Bp = wT + (size_t)nt * 64 * Cc;

  GEMM_PIPELINE(24, Cc,
                load_frag_h(A, Cc, kb),
                load_frag_h(A + 16 * Cc, Cc, kb))

  int lane  = lane_id();
  int col   = lane & 15;
  int rbase = (lane >> 4) << 3;
#pragma unroll
  for (int mi = 0; mi < 2; ++mi) {
#pragma unroll
    for (int j = 0; j < 4; ++j) {
#pragma unroll
      for (int i = 0; i < 8; ++i) {
        int t   = (mt0 + mi) * 16 + rbase + i;
        int o   = nt * 64 + j * 16 + col;
        int s   = o / Cc;
        int rem = o - s * Cc;
        int h   = rem >> 6;
        int d   = rem & 63;
        int bidx = t >> 10;
        int n    = t & 1023;
        size_t hd = (size_t)(bidx * Hh + h);
        float val = accs[mi][j][i];
        if (s == 0)      qh[(hd * Nn + n) * Dd + d] = (_Float16)(val * 0.125f); // fold 1/sqrt(D)
        else if (s == 1) kh[(hd * Nn + n) * Dd + d] = (_Float16)val;
        else             vT[(hd * Dd + d) * Nn + n] = (_Float16)val;
      }
    }
  }
}

// ---------------- scores: S = Q * K^T (per b,h), f32 out ----------------
__global__ __launch_bounds__(256) void scores_gemm(const _Float16* __restrict__ qh,
                                                   const _Float16* __restrict__ kh,
                                                   float* __restrict__ attn) {
  __shared__ _Float16 sB[2][64 * 32];
  int wid = (int)(threadIdx.x >> 5);
  int nt  = blockIdx.x & 15;                 // 16 key groups of 64
  int mtg = (blockIdx.x >> 4) & 3;           // 4 groups of 16 query tiles
  int bh  = blockIdx.x >> 6;                 // B*H = 96
  int mt0 = mtg * 16 + wid * 2;
  const _Float16* A  = qh + (size_t)bh * Nn * Dd + (size_t)mt0 * 16 * Dd;
  const _Float16* Bp = kh + (size_t)bh * Nn * Dd + (size_t)nt * 64 * Dd;

  GEMM_PIPELINE(2, Dd,
                load_frag_h(A, Dd, kb),
                load_frag_h(A + 16 * Dd, Dd, kb))

  int lane  = lane_id();
  int col   = lane & 15;
  int rbase = (lane >> 4) << 3;
#pragma unroll
  for (int mi = 0; mi < 2; ++mi) {
#pragma unroll
    for (int j = 0; j < 4; ++j) {
#pragma unroll
      for (int i = 0; i < 8; ++i) {
        int qrow = (mt0 + mi) * 16 + rbase + i;
        int kcol = nt * 64 + j * 16 + col;
        attn[((size_t)bh * Nn + qrow) * Nn + kcol] = accs[mi][j][i];
      }
    }
  }
}

// ---------------- softmax over rows of length 1024, in place ----------------
__global__ __launch_bounds__(256) void softmax_rows(float* __restrict__ attn) {
  int row = blockIdx.x * 8 + (threadIdx.x >> 5);  // B*H*N = 98304 rows
  float* p = attn + (size_t)row * Nn;
  int lane = lane_id();

  float vals[32];
#pragma unroll
  for (int i = 0; i < 8; ++i)
    *(float4*)(vals + i * 4) = *(const float4*)(p + i * 128 + lane * 4);

  float m = -1e30f;
#pragma unroll
  for (int i = 0; i < 32; ++i) m = fmaxf(m, vals[i]);
#pragma unroll
  for (int off = 16; off >= 1; off >>= 1) m = fmaxf(m, __shfl_xor(m, off, 32));

  float s = 0.0f;
#pragma unroll
  for (int i = 0; i < 32; ++i) { vals[i] = __expf(vals[i] - m); s += vals[i]; }
#pragma unroll
  for (int off = 16; off >= 1; off >>= 1) s += __shfl_xor(s, off, 32);

  float inv = 1.0f / s;
#pragma unroll
  for (int i = 0; i < 32; ++i) vals[i] *= inv;
#pragma unroll
  for (int i = 0; i < 8; ++i)
    *(float4*)(p + i * 128 + lane * 4) = *(const float4*)(vals + i * 4);
}

// ---------------- ctx = P * V : A from f32 attn, V panel async-staged ----------------
__global__ __launch_bounds__(256) void ctx_gemm(const float* __restrict__ attn,
                                                const _Float16* __restrict__ vT,
                                                _Float16* __restrict__ ctxh) {
  __shared__ _Float16 sB[2][64 * 32];
  int wid = (int)(threadIdx.x >> 5);
  int mtg = blockIdx.x & 3;                  // 4 groups of 16 query tiles
  int bh  = blockIdx.x >> 2;                 // 96
  int mt0 = mtg * 16 + wid * 2;
  const float*    A  = attn + (size_t)bh * Nn * Nn + (size_t)mt0 * 16 * Nn;
  const _Float16* Bp = vT   + (size_t)bh * Dd * Nn;   // 64 rows (d) x 1024 (keys)

  GEMM_PIPELINE(32, Nn,
                load_frag_f32(A, Nn, kb),
                load_frag_f32(A + 16 * Nn, Nn, kb))

  int bidx = bh / Hh;
  int h    = bh - bidx * Hh;
  int lane  = lane_id();
  int col   = lane & 15;
  int rbase = (lane >> 4) << 3;
#pragma unroll
  for (int mi = 0; mi < 2; ++mi) {
#pragma unroll
    for (int j = 0; j < 4; ++j) {
#pragma unroll
      for (int i = 0; i < 8; ++i) {
        int n = (mt0 + mi) * 16 + rbase + i;
        int d = j * 16 + col;
        ctxh[((size_t)(bidx * Nn + n)) * Cc + h * Dd + d] = (_Float16)accs[mi][j][i];
      }
    }
  }
}

// ---------------- out = ctx @ w_proj + b_proj ----------------
__global__ __launch_bounds__(256) void proj_gemm(const _Float16* __restrict__ ctxh,
                                                 const _Float16* __restrict__ wT,
                                                 const float* __restrict__ bias,
                                                 float* __restrict__ out) {
  __shared__ _Float16 sB[2][64 * 32];
  int wid = (int)(threadIdx.x >> 5);
  int nt  = blockIdx.x % 12;                 // 12 groups of 64 output cols
  int mtg = blockIdx.x / 12;                 // 32 groups of 16 M tiles
  int mt0 = mtg * 16 + wid * 2;
  const _Float16* A  = ctxh + (size_t)mt0 * 16 * Cc;
  const _Float16* Bp = wT   + (size_t)nt * 64 * Cc;

  GEMM_PIPELINE(24, Cc,
                load_frag_h(A, Cc, kb),
                load_frag_h(A + 16 * Cc, Cc, kb))

  int lane  = lane_id();
  int col   = lane & 15;
  int rbase = (lane >> 4) << 3;
#pragma unroll
  for (int mi = 0; mi < 2; ++mi) {
#pragma unroll
    for (int j = 0; j < 4; ++j) {
#pragma unroll
      for (int i = 0; i < 8; ++i) {
        int t = (mt0 + mi) * 16 + rbase + i;
        int o = nt * 64 + j * 16 + col;
        out[(size_t)t * Cc + o] = accs[mi][j][i] + bias[o];
      }
    }
  }
}

// ---------------- launcher ----------------
extern "C" void kernel_launch(void* const* d_in, const int* in_sizes, int n_in,
                              void* d_out, int out_size, void* d_ws, size_t ws_size,
                              hipStream_t stream) {
  const float* x      = (const float*)d_in[0];
  const float* w_qkv  = (const float*)d_in[1];
  const float* w_proj = (const float*)d_in[2];
  const float* b_proj = (const float*)d_in[3];

  float* out  = (float*)d_out;
  float* attn = out + (size_t)Bb * Nn * Cc;      // second tuple element

  _Float16* ws     = (_Float16*)d_ws;
  _Float16* xh     = ws + XH_OFF;
  _Float16* wqkvT  = ws + WQKVT_OFF;
  _Float16* wprojT = ws + WPROJT_OFF;
  _Float16* qh     = ws + QH_OFF;
  _Float16* kh     = ws + KH_OFF;
  _Float16* vT     = ws + VT_OFF;
  _Float16* ctxh   = ws + CTXH_OFF;

  cvt_f32_to_f16   <<<24576, 256, 0, stream>>>(x, xh, Bb * Nn * Cc);
  cvt_transpose_f16<<<6912,  256, 0, stream>>>(w_qkv,  wqkvT,  Cc, C3);
  cvt_transpose_f16<<<2304,  256, 0, stream>>>(w_proj, wprojT, Cc, Cc);

  qkv_gemm    <<<1152, 256, 0, stream>>>(xh, wqkvT, qh, kh, vT);
  scores_gemm <<<6144, 256, 0, stream>>>(qh, kh, attn);
  softmax_rows<<<12288, 256, 0, stream>>>(attn);
  ctx_gemm    <<<384,  256, 0, stream>>>(attn, vT, ctxh);
  proj_gemm   <<<384,  256, 0, stream>>>(ctxh, wprojT, b_proj, out);
}
